// GAT_69097433858681
// MI455X (gfx1250) — matrix operand order repaired and checked by reading, not compiled, not run on hardware
//
#include <hip/hip_runtime.h>
#include <hip/hip_bf16.h>
#include <math.h>

typedef __attribute__((ext_vector_type(16))) __bf16 v16bf;
typedef __attribute__((ext_vector_type(8)))  __bf16 v8bf;
typedef __attribute__((ext_vector_type(8)))  float  v8f;

#define LDSTRIDE 40   // bf16 elements per LDS row (80B): 16B-aligned, bank-skewed

// Async global->LDS path (ASYNCcnt-tracked) if this toolchain declares it.
#if defined(__gfx1250__) && \
    __has_builtin(__builtin_amdgcn_global_load_async_to_lds_b128) && \
    __has_builtin(__builtin_amdgcn_s_wait_asynccnt)
#define GAT_ASYNC 1
typedef int v4i_b128 __attribute__((vector_size(16)));
typedef __attribute__((address_space(1))) v4i_b128* gptr_b128;   // global src
typedef __attribute__((address_space(3))) v4i_b128* lptr_b128;   // LDS dst
#else
#define GAT_ASYNC 0
#endif

// ---------------------------------------------------------------------------
// Tiled bf16 WMMA GEMM:  Out[M,NCOL] = A[M,K] * W[K,NCOL]  (fp32 accumulate)
// 256 threads = 8 waves; each wave owns a 16-row slab of a 128-row tile.
// Double-buffered LDS; global loads for tile i+1 issued before the WMMA chain
// of tile i, LDS stores after it (one barrier per k-step).  All NT B-fragments
// are preloaded so WMMAs issue back-to-back with counted dscnt waits.
// Fragment layouts follow CDNA5 ISA 7.12.2 (wave32).
// ---------------------------------------------------------------------------
template<int K, int NCOL, bool ABF16>
__global__ __launch_bounds__(256) void gat_gemm(const void* __restrict__ Aglob,
                                                const float* __restrict__ Wglob,
                                                float* __restrict__ Out, int M)
{
    constexpr int STEPS  = K / 32;
    constexpr int NT     = NCOL / 16;       // 16x16 output tiles per wave
    constexpr int GROUPS = 256 / NCOL;
    constexpr int KPER   = 32 / GROUPS;

    __shared__ __bf16 Asl[2][128 * LDSTRIDE];
    __shared__ __bf16 Bsl[2][NCOL * LDSTRIDE];

    const int tid  = threadIdx.x;
    const int wave = tid >> 5;
    const int lane = tid & 31;
    const int half = lane >> 4;
    const int l16  = lane & 15;
    const int rowBase = blockIdx.x * 128;

    v8f acc[NT];
#pragma unroll
    for (int i = 0; i < NT; ++i) {
        v8f z = {0.f,0.f,0.f,0.f,0.f,0.f,0.f,0.f};
        acc[i] = z;
    }

    // A staging: 2 threads per row, 16 k-elements each
    const int r  = tid >> 1;
    const int kk = (tid & 1) * 16;
    int row = rowBase + r;
    if (row >= M) row = M - 1;              // clamp (stores are guarded)

    // B staging: NCOL columns x GROUPS k-groups
    const int bn = tid % NCOL;
    const int bk = (tid / NCOL) * KPER;

    alignas(16) float  aF[16];              // fp32-A path staging regs
    alignas(16) __bf16 aH[16];              // bf16-A sync path staging regs
    float bF[KPER];

    auto loadA = [&](int k0, int dbuf) {
        if constexpr (ABF16) {
            const __bf16* Ag = (const __bf16*)Aglob + (size_t)row * K + k0 + kk;
#if GAT_ASYNC
            // DMA straight into LDS, no VGPR round trip (ASYNCcnt-tracked)
            __builtin_amdgcn_global_load_async_to_lds_b128(
                (gptr_b128)Ag,
                (lptr_b128)&Asl[dbuf][r * LDSTRIDE + kk], 0, 0);
            __builtin_amdgcn_global_load_async_to_lds_b128(
                (gptr_b128)(Ag + 8),
                (lptr_b128)&Asl[dbuf][r * LDSTRIDE + kk + 8], 0, 0);
#else
            (void)dbuf;
            *(v8bf*)&aH[0] = *(const v8bf*)Ag;
            *(v8bf*)&aH[8] = *(const v8bf*)(Ag + 8);
#endif
        } else {
            (void)dbuf;
            const float* Ag = (const float*)Aglob + (size_t)row * K + k0 + kk;
            if (k0 + 64 < K) __builtin_prefetch(Ag + 64, 0, 1);  // global_prefetch_b8
#pragma unroll
            for (int i = 0; i < 16; i += 4)
                *(float4*)&aF[i] = *(const float4*)(Ag + i);
        }
    };
    auto storeA = [&](int dbuf) {
        if constexpr (ABF16) {
#if !GAT_ASYNC
            *(v8bf*)&Asl[dbuf][r * LDSTRIDE + kk]     = *(v8bf*)&aH[0];
            *(v8bf*)&Asl[dbuf][r * LDSTRIDE + kk + 8] = *(v8bf*)&aH[8];
#else
            (void)dbuf;
#endif
        } else {
            alignas(16) __bf16 t[16];
#pragma unroll
            for (int i = 0; i < 16; ++i) t[i] = (__bf16)aF[i];
            *(v8bf*)&Asl[dbuf][r * LDSTRIDE + kk]     = *(v8bf*)&t[0];
            *(v8bf*)&Asl[dbuf][r * LDSTRIDE + kk + 8] = *(v8bf*)&t[8];
        }
    };
    auto loadB = [&](int k0) {
#pragma unroll
        for (int i = 0; i < KPER; ++i)
            bF[i] = Wglob[(size_t)(k0 + bk + i) * NCOL + bn];
    };
    auto storeB = [&](int dbuf) {
        alignas(16) __bf16 t[KPER];
#pragma unroll
        for (int i = 0; i < KPER; ++i) t[i] = (__bf16)bF[i];
#pragma unroll
        for (int i = 0; i < KPER; i += 8)
            *(v8bf*)&Bsl[dbuf][bn * LDSTRIDE + bk + i] = *(v8bf*)&t[i];
    };

    // prologue: stage tile 0 into buffer 0
    loadA(0, 0); loadB(0); storeA(0); storeB(0);

    for (int i = 0; i < STEPS; ++i) {
        const int buf = i & 1;
#if GAT_ASYNC
        if constexpr (ABF16) __builtin_amdgcn_s_wait_asynccnt(0);
#endif
        __syncthreads();                    // publish buffer `buf`

        const bool more = (i + 1 < STEPS);
        if (more) { loadA((i + 1) * 32, buf ^ 1); loadB((i + 1) * 32); }

        // A fragment (16x32): lanes 0-15 hold K 0-7 & 16-23, lanes 16-31 K 8-15 & 24-31
        const int arow = wave * 16 + l16;
        v8bf alo = *(const v8bf*)&Asl[buf][arow * LDSTRIDE + half * 8];
        v8bf ahi = *(const v8bf*)&Asl[buf][arow * LDSTRIDE + 16 + half * 8];
        v16bf afrag = __builtin_shufflevector(alo, ahi,
                        0,1,2,3,4,5,6,7,8,9,10,11,12,13,14,15);

        // preload all B fragments (32x16 each: column n = lane%16, K = half*8+e, 16+half*8+(e-8))
        v16bf bfr[NT];
#pragma unroll
        for (int ct = 0; ct < NT; ++ct) {
            const int n = ct * 16 + l16;
            v8bf blo = *(const v8bf*)&Bsl[buf][n * LDSTRIDE + half * 8];
            v8bf bhi = *(const v8bf*)&Bsl[buf][n * LDSTRIDE + 16 + half * 8];
            bfr[ct] = __builtin_shufflevector(blo, bhi,
                        0,1,2,3,4,5,6,7,8,9,10,11,12,13,14,15);
        }
#pragma unroll
        for (int ct = 0; ct < NT; ++ct)
            acc[ct] = __builtin_amdgcn_wmma_f32_16x16x32_bf16(
                          false, afrag, false, bfr[ct], (short)0, acc[ct], false, false);

        if (more) { storeA(buf ^ 1); storeB(buf ^ 1); }
    }

    // D layout: lane -> column n = ct*16 + l16; VGPR rr -> row m = 8*half + rr
#pragma unroll
    for (int ct = 0; ct < NT; ++ct) {
#pragma unroll
        for (int rr = 0; rr < 8; ++rr) {
            int orow = rowBase + wave * 16 + half * 8 + rr;
            if (orow < M) Out[(size_t)orow * NCOL + ct * 16 + l16] = acc[ct][rr];
        }
    }
}

// ---------------------------------------------------------------------------
// Per-node attention logits: alpha_src/dst[n,h] = <h[n,h,:], a_src/dst[h,:]>
// ---------------------------------------------------------------------------
template<int H, int C>
__global__ __launch_bounds__(256) void node_alpha(const float* __restrict__ h,
                                                  const float* __restrict__ a_src,
                                                  const float* __restrict__ a_dst,
                                                  float* __restrict__ asrc,
                                                  float* __restrict__ adst, int N)
{
    int t = blockIdx.x * blockDim.x + threadIdx.x;
    if (t >= N * H) return;
    int n = t / H, hd = t % H;
    const float* hp = h + (size_t)n * H * C + hd * C;
    float ss = 0.f, sd = 0.f;
#pragma unroll
    for (int c = 0; c < C; c += 4) {
        float4 v  = *(const float4*)(hp + c);
        float4 as = *(const float4*)(a_src + hd * C + c);
        float4 ad = *(const float4*)(a_dst + hd * C + c);
        ss += v.x*as.x + v.y*as.y + v.z*as.z + v.w*as.w;
        sd += v.x*ad.x + v.y*ad.y + v.z*ad.z + v.w*ad.w;
    }
    asrc[t] = ss;
    adst[t] = sd;
}

__device__ __forceinline__ void edge_pair(const int* __restrict__ ei, int e, int E,
                                          int& s, int& d)
{
    if (e < E) { s = ei[e]; d = ei[E + e]; }
    else       { s = e - E; d = e - E; }       // self-loop
}

__device__ __forceinline__ float lrelu(float s) { return s > 0.f ? s : 0.2f * s; }

// segment max over dst (float atomic-max via signed/unsigned int trick; init -inf)
template<int H>
__global__ __launch_bounds__(256) void edge_max(const int* __restrict__ ei, int E, int N,
                                                const float* __restrict__ asrc,
                                                const float* __restrict__ adst,
                                                float* __restrict__ m)
{
    long long t = (long long)blockIdx.x * blockDim.x + threadIdx.x;
    if (t >= (long long)(E + N) * H) return;
    int e = (int)(t / H), h = (int)(t % H);
    int s, d; edge_pair(ei, e, E, s, d);
    float v = lrelu(asrc[(size_t)s * H + h] + adst[(size_t)d * H + h]);
    float* addr = &m[(size_t)d * H + h];
    if (v >= 0.f) atomicMax((int*)addr, __float_as_int(v));
    else          atomicMin((unsigned int*)addr, __float_as_uint(v));
}

template<int H>
__global__ __launch_bounds__(256) void edge_sum(const int* __restrict__ ei, int E, int N,
                                                const float* __restrict__ asrc,
                                                const float* __restrict__ adst,
                                                const float* __restrict__ m,
                                                float* __restrict__ denom)
{
    long long t = (long long)blockIdx.x * blockDim.x + threadIdx.x;
    if (t >= (long long)(E + N) * H) return;
    int e = (int)(t / H), h = (int)(t % H);
    int s, d; edge_pair(ei, e, E, s, d);
    float v = lrelu(asrc[(size_t)s * H + h] + adst[(size_t)d * H + h]);
    float ex = __expf(v - m[(size_t)d * H + h]);
    atomicAdd(&denom[(size_t)d * H + h], ex);
}

// message aggregation: each thread handles 4 channels of one edge
template<int H, int C>
__global__ __launch_bounds__(256) void edge_agg(const int* __restrict__ ei, int E, int N,
                                                const float* __restrict__ asrc,
                                                const float* __restrict__ adst,
                                                const float* __restrict__ m,
                                                const float* __restrict__ denom,
                                                const float* __restrict__ h,
                                                float* __restrict__ agg)
{
    constexpr int HC  = H * C;
    constexpr int TPE = HC / 4;
    long long t = (long long)blockIdx.x * blockDim.x + threadIdx.x;
    if (t >= (long long)(E + N) * TPE) return;
    int e = (int)(t / TPE);
    int j = (int)(t % TPE);
    int hd = (j * 4) / C;
    int s, d; edge_pair(ei, e, E, s, d);
    float v = lrelu(asrc[(size_t)s * H + hd] + adst[(size_t)d * H + hd]);
    float alpha = __expf(v - m[(size_t)d * H + hd]) / denom[(size_t)d * H + hd];
    float4 f = *(const float4*)(h + (size_t)s * HC + j * 4);
    float* o = agg + (size_t)d * HC + j * 4;
    atomicAdd(o + 0, f.x * alpha);
    atomicAdd(o + 1, f.y * alpha);
    atomicAdd(o + 2, f.z * alpha);
    atomicAdd(o + 3, f.w * alpha);
}

__global__ __launch_bounds__(256) void bias_elu_bf16(const float* __restrict__ agg,
                                                     const float* __restrict__ bias,
                                                     __bf16* __restrict__ out,
                                                     int total, int C)
{
    int t = blockIdx.x * blockDim.x + threadIdx.x;
    if (t >= total) return;
    float v = agg[t] + bias[t % C];
    v = v > 0.f ? v : (__expf(v) - 1.0f);    // ELU
    out[t] = (__bf16)v;
}

__global__ __launch_bounds__(256) void fill_f32(float* __restrict__ p, float v, long long n)
{
    long long t = (long long)blockIdx.x * blockDim.x + threadIdx.x;
    if (t < n) p[t] = v;
}

__global__ __launch_bounds__(256) void init_out_bias(float* __restrict__ out,
                                                     const float* __restrict__ b,
                                                     int N, int C)
{
    int t = blockIdx.x * blockDim.x + threadIdx.x;
    if (t < N * C) out[t] = b[t % C];
}

// ---------------------------------------------------------------------------
extern "C" void kernel_launch(void* const* d_in, const int* in_sizes, int n_in,
                              void* d_out, int out_size, void* d_ws, size_t ws_size,
                              hipStream_t stream)
{
    constexpr int N = 100000, E = 1600000;
    constexpr int INC = 512, H1 = 8, C1 = 16, HC1 = 128, H2 = 1, C2 = 64;
    (void)in_sizes; (void)n_in; (void)out_size; (void)ws_size;

    const float* x    = (const float*)d_in[0];
    const int*   ei   = (const int*)  d_in[1];
    const float* W1   = (const float*)d_in[2];
    const float* a_s1 = (const float*)d_in[3];
    const float* a_d1 = (const float*)d_in[4];
    const float* b1   = (const float*)d_in[5];
    const float* W2   = (const float*)d_in[6];
    const float* a_s2 = (const float*)d_in[7];
    const float* a_d2 = (const float*)d_in[8];
    const float* b2   = (const float*)d_in[9];
    float* out = (float*)d_out;

    // workspace carve-up (~168 MB) — features + aggregates stay L2-resident
    char* w = (char*)d_ws;
    auto take = [&](size_t bytes) -> char* {
        char* p = w; w += (bytes + 255) & ~size_t(255); return p;
    };
    float*  h1   = (float*) take((size_t)N * HC1 * 4);
    float*  agg1 = (float*) take((size_t)N * HC1 * 4);
    float*  h2   = (float*) take((size_t)N * C2  * 4);
    __bf16* hh   = (__bf16*)take((size_t)N * HC1 * 2);
    float*  as1  = (float*) take((size_t)N * H1 * 4);
    float*  ad1  = (float*) take((size_t)N * H1 * 4);
    float*  m1   = (float*) take((size_t)N * H1 * 4);
    float*  dn1  = (float*) take((size_t)N * H1 * 4);
    float*  as2  = (float*) take((size_t)N * 4);
    float*  ad2  = (float*) take((size_t)N * 4);
    float*  m2   = (float*) take((size_t)N * 4);
    float*  dn2  = (float*) take((size_t)N * 4);

    const int EP = E + N;
    dim3 blk(256);
    auto g = [](long long work) { return dim3((unsigned)((work + 255) / 256)); };

    // ---------------- layer 1 ----------------
    gat_gemm<INC, HC1, false><<<dim3((N + 127) / 128), blk, 0, stream>>>(x, W1, h1, N);
    node_alpha<H1, C1><<<g((long long)N * H1), blk, 0, stream>>>(h1, a_s1, a_d1, as1, ad1, N);
    fill_f32<<<g((long long)N * H1), blk, 0, stream>>>(m1, -INFINITY, (long long)N * H1);
    fill_f32<<<g((long long)N * H1), blk, 0, stream>>>(dn1, 0.f, (long long)N * H1);
    fill_f32<<<g((long long)N * HC1), blk, 0, stream>>>(agg1, 0.f, (long long)N * HC1);
    edge_max<H1><<<g((long long)EP * H1), blk, 0, stream>>>(ei, E, N, as1, ad1, m1);
    edge_sum<H1><<<g((long long)EP * H1), blk, 0, stream>>>(ei, E, N, as1, ad1, m1, dn1);
    edge_agg<H1, C1><<<g((long long)EP * (H1 * C1 / 4)), blk, 0, stream>>>(
        ei, E, N, as1, ad1, m1, dn1, h1, agg1);
    bias_elu_bf16<<<g((long long)N * HC1), blk, 0, stream>>>(agg1, b1, hh, N * HC1, HC1);

    // ---------------- layer 2 ----------------
    gat_gemm<HC1, C2, true><<<dim3((N + 127) / 128), blk, 0, stream>>>(hh, W2, h2, N);
    node_alpha<H2, C2><<<g(N), blk, 0, stream>>>(h2, a_s2, a_d2, as2, ad2, N);
    fill_f32<<<g(N), blk, 0, stream>>>(m2, -INFINITY, N);
    fill_f32<<<g(N), blk, 0, stream>>>(dn2, 0.f, N);
    init_out_bias<<<g((long long)N * C2), blk, 0, stream>>>(out, b2, N, C2);
    edge_max<H2><<<g(EP), blk, 0, stream>>>(ei, E, N, as2, ad2, m2);
    edge_sum<H2><<<g(EP), blk, 0, stream>>>(ei, E, N, as2, ad2, m2, dn2);
    edge_agg<H2, C2><<<g((long long)EP * (C2 / 4)), blk, 0, stream>>>(
        ei, E, N, as2, ad2, m2, dn2, h2, out);
}
